// MultiheadAttention_37288906064443
// MI455X (gfx1250) — matrix operand order repaired and checked by reading, not compiled, not run on hardware
//
#include <hip/hip_runtime.h>
#include <hip/hip_bf16.h>

#define EDIM 1024
#define HEADS 16
#define DHEAD 64
#define BATCH 8
#define SEQ   1024
#define MTOT  (BATCH * SEQ)      // 8192
#define QKVN  (3 * EDIM)         // 3072

typedef __attribute__((ext_vector_type(16))) __bf16 v16bf;
typedef __attribute__((ext_vector_type(8)))  __bf16 v8bf;
typedef __attribute__((ext_vector_type(4)))  __bf16 v4bf;
typedef __attribute__((ext_vector_type(8)))  float  v8f;
typedef __attribute__((ext_vector_type(4)))  int    v4i;

#define AS1 __attribute__((address_space(1)))
#define AS3 __attribute__((address_space(3)))

// ---------------- fragment loaders (CDNA5 ISA 7.12.2 layouts) ----------------

// 16-bit A-matrix 16x32: lanes 0-15 (M=lane): K = k0+0..7 , k0+16..23
//                        lanes 16-31 (M=lane-16): K = k0+8..15, k0+24..31
__device__ __forceinline__ v16bf load_a_frag(const __bf16* __restrict__ base,
                                             int row, int stride, int k0) {
  int lane = threadIdx.x & 31;
  const __bf16* p = base + (size_t)row * stride + k0 + ((lane & 16) ? 8 : 0);
  union { v16bf v; v8bf h[2]; } u;
  u.h[0] = *(const v8bf*)(p);
  u.h[1] = *(const v8bf*)(p + 16);
  return u.v;
}

// 16-bit B-matrix 32x16: lanes 0-15 (N=lane): K = k0+0..7 , k0+16..23
//                        lanes 16-31 (N=lane-16): K = k0+8..15, k0+24..31
__device__ __forceinline__ v16bf load_b_frag(const __bf16* __restrict__ base,
                                             int col, int stride, int k0) {
  int lane = threadIdx.x & 31;
  const __bf16* p = base + (size_t)col * stride + k0 + ((lane & 16) ? 8 : 0);
  union { v16bf v; v8bf h[2]; } u;
  u.h[0] = *(const v8bf*)(p);
  u.h[1] = *(const v8bf*)(p + 16);
  return u.v;
}

__device__ __forceinline__ v8f wmma_bf16(v16bf a, v16bf b, v8f c) {
  return __builtin_amdgcn_wmma_f32_16x16x32_bf16(false, a, false, b,
                                                 (short)0, c, false, false);
}

// --------- async global->LDS staging (CDNA5 ASYNCcnt path, with fallback) ----

__device__ __forceinline__ void stage16(const __bf16* g, __bf16* l) {
#if __has_builtin(__builtin_amdgcn_global_load_async_to_lds_b128)
  __builtin_amdgcn_global_load_async_to_lds_b128(
      (AS1 v4i*)(AS1 void*)g, (AS3 v4i*)(AS3 void*)l, 0, 0);
#else
  *(v8bf*)l = *(const v8bf*)g;
#endif
}

__device__ __forceinline__ void wait_async4() {
#if __has_builtin(__builtin_amdgcn_s_wait_asynccnt)
  __builtin_amdgcn_s_wait_asynccnt(4);   // oldest tile's 4 copies complete
#endif
}

__device__ __forceinline__ void wait_async0() {
#if __has_builtin(__builtin_amdgcn_s_wait_asynccnt)
  __builtin_amdgcn_s_wait_asynccnt(0);
#endif
}

// ---------------- fp32 -> bf16 conversion ----------------

__global__ void cvt_f32_bf16(const float4* __restrict__ src,
                             v4bf* __restrict__ dst, int n4) {
  int i = blockIdx.x * blockDim.x + threadIdx.x;
  if (i < n4) {
    float4 f = src[i];
    v4bf o;
    o.x = (__bf16)f.x; o.y = (__bf16)f.y; o.z = (__bf16)f.z; o.w = (__bf16)f.w;
    dst[i] = o;
  }
}

// ---------------- QKV projection: qkv = x @ w_in^T + b_in ----------------
// Wave tile: 32(M) x 64(N). Writes q[BH,S,D], k[BH,S,D], v^T[BH,D,S] in bf16.

__global__ void qkv_gemm(const __bf16* __restrict__ xb,    // [8192, 1024]
                         const __bf16* __restrict__ wb,    // [3072, 1024]
                         const float*  __restrict__ bias,  // [3072]
                         __bf16* __restrict__ q,
                         __bf16* __restrict__ k,
                         __bf16* __restrict__ vt) {
  int lane = threadIdx.x & 31;
  int wid  = threadIdx.x >> 5;
  int wm = wid & 1, wn = wid >> 1;                // 2 x 4 waves
  int m0 = blockIdx.x * 64  + wm * 32;
  int n0 = blockIdx.y * 256 + wn * 64;

  v8f acc[2][4] = {};
  for (int kk = 0; kk < EDIM; kk += 32) {
    v16bf a[2], bf[4];
#pragma unroll
    for (int i = 0; i < 2; ++i)
      a[i] = load_a_frag(xb, m0 + i * 16 + (lane & 15), EDIM, kk);
#pragma unroll
    for (int j = 0; j < 4; ++j)
      bf[j] = load_b_frag(wb, n0 + j * 16 + (lane & 15), EDIM, kk);
#pragma unroll
    for (int i = 0; i < 2; ++i)
#pragma unroll
      for (int j = 0; j < 4; ++j)
        acc[i][j] = wmma_bf16(a[i], bf[j], acc[i][j]);
  }

#pragma unroll
  for (int i = 0; i < 2; ++i) {
#pragma unroll
    for (int j = 0; j < 4; ++j) {
      int f   = n0 + j * 16 + (lane & 15);
      float bv = bias[f];
      int h   = f / (3 * DHEAD);
      int c   = f % (3 * DHEAD);
      int grp = c >> 6;
      int d   = c & 63;
#pragma unroll
      for (int r = 0; r < 8; ++r) {
        int m = m0 + i * 16 + ((lane & 16) ? 8 : 0) + r;   // b*S + s
        int b = m >> 10, s = m & (SEQ - 1);
        __bf16 o = (__bf16)(acc[i][j][r] + bv);
        size_t bh = (size_t)(b * HEADS + h);
        if (grp == 0)      q [(bh * SEQ + s) * DHEAD + d] = o;
        else if (grp == 1) k [(bh * SEQ + s) * DHEAD + d] = o;
        else               vt[(bh * DHEAD + d) * SEQ + s] = o;
      }
    }
  }
}

// ---------------- flash attention ----------------
// One wave per 16-query block; 8 waves/workgroup share one (b,h) head. K/V
// tiles are staged into LDS once per workgroup via async global->LDS copies,
// DOUBLE-BUFFERED on ASYNCcnt: tile i+1's copies are issued before tile i is
// consumed; s_wait_asynccnt(4) retires the 4 oldest (in-order) copies.
// logits^T = K_tile @ Q^T so the exp'd tile lands directly in A-fragment
// layout for P @ V (no LDS transpose). Online softmax, fp32 accumulators.

__global__ void flash_attn(const __bf16* __restrict__ qg,  // [BH, S, D]
                           const __bf16* __restrict__ kg,  // [BH, S, D]
                           const __bf16* __restrict__ vg,  // [BH, D, S]
                           __bf16* __restrict__ val) {     // [B, S, H, D]
  __shared__ alignas(16) __bf16 sK[2][64 * 64];   // [buf][key][d]   16 KB
  __shared__ alignas(16) __bf16 sV[2][64 * 64];   // [buf][d][key]   16 KB

  int tid  = threadIdx.x;
  int lane = tid & 31;
  int wid  = tid >> 5;
  int w  = blockIdx.x * 8 + wid;       // 0..8191
  int bh = w >> 6;                     // same for all 8 waves of the block
  int qb = (w & 63) * 16;              // query-row base
  const __bf16* qp = qg + (size_t)bh * SEQ * DHEAD;
  const __bf16* kp = kg + (size_t)bh * SEQ * DHEAD;
  const __bf16* vp = vg + (size_t)bh * DHEAD * SEQ;

  int st_r    = tid >> 3;              // staging row 0..31 (+32 for 2nd chunk)
  int st_c8   = (tid & 7) * 8;         // staging 8-element chunk

  // prologue: stage tile 0 into buffer 0 (4 async b128 copies per thread)
#pragma unroll
  for (int c = 0; c < 2; ++c) {
    int r = st_r + c * 32;
    stage16(kp + (size_t)r * DHEAD + st_c8, &sK[0][r * 64 + st_c8]);
    stage16(vp + (size_t)r * SEQ + st_c8,   &sV[0][r * 64 + st_c8]);
  }

  // Q as B-operand of K @ Q^T: column n = query row, K-dim = d
  v16bf qf[2];
#pragma unroll
  for (int i = 0; i < 2; ++i)
    qf[i] = load_b_frag(qp, qb + (lane & 15), DHEAD, i * 32);

  v8f acc[4] = {};                     // out tile 16q x 64d
  float m_run = -1e30f, l_run = 0.0f;
  const float scale = 0.125f;          // 1/sqrt(64)

  for (int kb = 0; kb < SEQ; kb += 64) {
    int cur = (kb >> 6) & 1;
    // prefetch next tile into the other buffer (consumed last iteration,
    // protected by the end-of-iteration barrier)
    if (kb + 64 < SEQ) {
      int nxt = cur ^ 1;
#pragma unroll
      for (int c = 0; c < 2; ++c) {
        int r = st_r + c * 32;
        stage16(kp + (size_t)(kb + 64 + r) * DHEAD + st_c8,
                &sK[nxt][r * 64 + st_c8]);
        stage16(vp + (size_t)r * SEQ + (kb + 64) + st_c8,
                &sV[nxt][r * 64 + st_c8]);
      }
      wait_async4();                   // tile `cur` (oldest 4 copies) landed
    } else {
      wait_async0();
    }
    __syncthreads();

    const __bf16* cK = sK[cur];
    const __bf16* cV = sV[cur];

    // logits^T tiles: ct[t] covers keys kb+16t .. kb+16t+15
    v8f ct[4];
#pragma unroll
    for (int t = 0; t < 4; ++t) {
      v16bf ka0 = load_a_frag(cK, t * 16 + (lane & 15), 64, 0);
      v16bf ka1 = load_a_frag(cK, t * 16 + (lane & 15), 64, 32);
      v8f c = {};
      c = wmma_bf16(ka0, qf[0], c);
      c = wmma_bf16(ka1, qf[1], c);
      ct[t] = c;
    }
    float mt = -1e30f;
#pragma unroll
    for (int t = 0; t < 4; ++t)
#pragma unroll
      for (int r = 0; r < 8; ++r) {
        ct[t][r] *= scale;
        mt = fmaxf(mt, ct[t][r]);
      }
    mt = fmaxf(mt, __shfl_xor(mt, 16, 32));   // stats per q = lane&15
    float m_new = fmaxf(m_run, mt);
    float alpha = __expf(m_run - m_new);
    float ssum = 0.0f;
    union { v16bf v; v8bf h[2]; } pa[2];      // P tiles in A-fragment layout
#pragma unroll
    for (int t = 0; t < 4; ++t)
#pragma unroll
      for (int r = 0; r < 8; ++r) {
        float p = __expf(ct[t][r] - m_new);
        ssum += p;
        pa[t >> 1].h[t & 1][r] = (__bf16)p;
      }
    ssum += __shfl_xor(ssum, 16, 32);
    l_run = l_run * alpha + ssum;
    m_run = m_new;

    // broadcast per-q alpha into C-row space: row r <-> q = (lane<16 ? r : 8+r)
    float ar[8];
#pragma unroll
    for (int r = 0; r < 8; ++r)
      ar[r] = __shfl(alpha, ((lane & 16) ? 8 : 0) + r, 32);

#pragma unroll
    for (int j = 0; j < 4; ++j) {
#pragma unroll
      for (int r = 0; r < 8; ++r) acc[j][r] *= ar[r];
      v16bf vf0 = load_b_frag(cV, j * 16 + (lane & 15), 64, 0);
      v16bf vf1 = load_b_frag(cV, j * 16 + (lane & 15), 64, 32);
      acc[j] = wmma_bf16(pa[0].v, vf0, acc[j]);
      acc[j] = wmma_bf16(pa[1].v, vf1, acc[j]);
    }
    __syncthreads();                   // tile consumed; safe to overwrite
  }

  float linv = 1.0f / l_run;
  float lr[8];
#pragma unroll
  for (int r = 0; r < 8; ++r)
    lr[r] = __shfl(linv, ((lane & 16) ? 8 : 0) + r, 32);

  int b = bh >> 4, h = bh & 15;
#pragma unroll
  for (int j = 0; j < 4; ++j) {
    int d = j * 16 + (lane & 15);
#pragma unroll
    for (int r = 0; r < 8; ++r) {
      int s = qb + ((lane & 16) ? 8 : 0) + r;
      val[(((size_t)b * SEQ + s) * HEADS + h) * DHEAD + d] =
          (__bf16)(acc[j][r] * lr[r]);
    }
  }
}

// ---------------- output projection: out = values @ w_out^T + b_out ----------

__global__ void out_gemm(const __bf16* __restrict__ vb,   // [8192, 1024]
                         const __bf16* __restrict__ wb,   // [1024, 1024]
                         const float*  __restrict__ bias, // [1024]
                         float* __restrict__ out) {       // [8192, 1024] fp32
  int lane = threadIdx.x & 31;
  int wid  = threadIdx.x >> 5;
  int wm = wid & 1, wn = wid >> 1;
  int m0 = blockIdx.x * 64  + wm * 32;
  int n0 = blockIdx.y * 256 + wn * 64;

  v8f acc[2][4] = {};
  for (int kk = 0; kk < EDIM; kk += 32) {
    v16bf a[2], bf[4];
#pragma unroll
    for (int i = 0; i < 2; ++i)
      a[i] = load_a_frag(vb, m0 + i * 16 + (lane & 15), EDIM, kk);
#pragma unroll
    for (int j = 0; j < 4; ++j)
      bf[j] = load_b_frag(wb, n0 + j * 16 + (lane & 15), EDIM, kk);
#pragma unroll
    for (int i = 0; i < 2; ++i)
#pragma unroll
      for (int j = 0; j < 4; ++j)
        acc[i][j] = wmma_bf16(a[i], bf[j], acc[i][j]);
  }

#pragma unroll
  for (int i = 0; i < 2; ++i) {
#pragma unroll
    for (int j = 0; j < 4; ++j) {
      int f = n0 + j * 16 + (lane & 15);
      float bv = bias[f];
#pragma unroll
      for (int r = 0; r < 8; ++r) {
        int m = m0 + i * 16 + ((lane & 16) ? 8 : 0) + r;
        out[(size_t)m * EDIM + f] = acc[i][j][r] + bv;
      }
    }
  }
}

// ---------------- host-side orchestration ----------------

extern "C" void kernel_launch(void* const* d_in, const int* in_sizes, int n_in,
                              void* d_out, int out_size, void* d_ws, size_t ws_size,
                              hipStream_t stream) {
  const float* x     = (const float*)d_in[0];   // [8, 1024, 1024]
  const float* w_in  = (const float*)d_in[1];   // [3072, 1024]
  const float* b_in  = (const float*)d_in[2];   // [3072]
  const float* w_out = (const float*)d_in[3];   // [1024, 1024]
  const float* b_out = (const float*)d_in[4];   // [1024]
  float* out = (float*)d_out;

  char* ws = (char*)d_ws;
  const size_t OFF_X    = 0;                        // 16 MB (reused for values)
  const size_t OFF_WIN  = 16777216;                 //  6 MB
  const size_t OFF_WOUT = 23068672;                 //  2 MB
  const size_t OFF_Q    = 25165824;                 // 16 MB
  const size_t OFF_K    = 41943040;                 // 16 MB
  const size_t OFF_V    = 58720256;                 // 16 MB  (total 72 MB)

  __bf16* xb   = (__bf16*)(ws + OFF_X);
  __bf16* winb = (__bf16*)(ws + OFF_WIN);
  __bf16* woutb= (__bf16*)(ws + OFF_WOUT);
  __bf16* qb   = (__bf16*)(ws + OFF_Q);
  __bf16* kb   = (__bf16*)(ws + OFF_K);
  __bf16* vtb  = (__bf16*)(ws + OFF_V);
  __bf16* valb = xb;  // values tensor reuses x_bf16 region (x dead by then)

  // fp32 -> bf16
  {
    int n4 = (MTOT * EDIM) / 4;
    cvt_f32_bf16<<<(n4 + 255) / 256, 256, 0, stream>>>((const float4*)x, (v4bf*)xb, n4);
    n4 = (QKVN * EDIM) / 4;
    cvt_f32_bf16<<<(n4 + 255) / 256, 256, 0, stream>>>((const float4*)w_in, (v4bf*)winb, n4);
    n4 = (EDIM * EDIM) / 4;
    cvt_f32_bf16<<<(n4 + 255) / 256, 256, 0, stream>>>((const float4*)w_out, (v4bf*)woutb, n4);
  }

  // QKV projection: [8192,3072] = [8192,1024] @ [3072,1024]^T
  qkv_gemm<<<dim3(MTOT / 64, QKVN / 256), 256, 0, stream>>>(xb, winb, b_in, qb, kb, vtb);

  // attention: 8192 query blocks of 16 rows, 8 waves/block (one head per block)
  flash_attn<<<(BATCH * HEADS * (SEQ / 16)) / 8, 256, 0, stream>>>(qb, kb, vtb, valb);

  // output projection: [8192,1024] = [8192,1024] @ [1024,1024]^T
  out_gemm<<<dim3(MTOT / 64, EDIM / 256), 256, 0, stream>>>(valb, woutb, b_out, out);
}
